// CrossAttention_37082747633821
// MI455X (gfx1250) — matrix operand order repaired and checked by reading, not compiled, not run on hardware
//
#include <hip/hip_runtime.h>
#include <hip/hip_bf16.h>
#include <stdint.h>

// ---------------------------------------------------------------------------
// Cross-attention for MI455X (gfx1250, wave32, WMMA).
// Pipeline: proj_q -> proj_kv -> flash-attention -> out-proj.
// All GEMMs use v_wmma_f32_16x16x32_bf16 (bf16 in, fp32 accumulate).
// Workspace layout (bf16): Q[B,H,L,64] | K[B,H,L,64] | Vt[B,H,64,L] | Y[B*L,D]
// = 4 x 16.78 MB = ~67.1 MB required in d_ws.
// ---------------------------------------------------------------------------

typedef __attribute__((ext_vector_type(16))) __bf16 v16bf;
typedef __attribute__((ext_vector_type(8)))  float  v8f;
typedef int v4i __attribute__((vector_size(16)));   // <4 x i32> for async-LDS builtin
typedef unsigned int u32;

#define DMODEL 1024
#define NHEADS 16
#define DHEAD  64
#define SEQL   2048
#define BATCH  4
#define WPAD   40      // W-tile LDS row stride (80B: 16B-aligned, conflict-free)
#define KLP    72      // K-tile LDS row stride (144B)
#define VLP    40      // V-tile LDS row stride (80B)

#if defined(__has_builtin)
#  if __has_builtin(__builtin_amdgcn_global_load_async_to_lds_b128)
#    define HAS_ASYNC_LDS 1
#  else
#    define HAS_ASYNC_LDS 0
#  endif
#  if __has_builtin(__builtin_amdgcn_s_wait_asynccnt)
#    define HAS_WAIT_ASYNC 1
#  else
#    define HAS_WAIT_ASYNC 0
#  endif
#else
#  define HAS_ASYNC_LDS 0
#  define HAS_WAIT_ASYNC 0
#endif

union VB16 { v16bf v; uint4 u[2]; };

static __device__ __forceinline__ v16bf load16bf(const __bf16* p) {
  VB16 r;
  r.u[0] = *(const uint4*)(p);
  r.u[1] = *(const uint4*)(p + 8);
  return r.v;
}

// A-operand (16-bit, 16x32): two 8-element runs at +off and +off+16.
static __device__ __forceinline__ v16bf loadA_bf16(const __bf16* p, int off) {
  VB16 r;
  r.u[0] = *(const uint4*)(p + off);
  r.u[1] = *(const uint4*)(p + off + 16);
  return r.v;
}

static __device__ __forceinline__ v16bf loadA_f32(const float* p, int off) {
  const float4 a = *(const float4*)(p + off);
  const float4 b = *(const float4*)(p + off + 4);
  const float4 c = *(const float4*)(p + off + 16);
  const float4 d = *(const float4*)(p + off + 20);
  v16bf v;
  v[0]=(__bf16)a.x;  v[1]=(__bf16)a.y;  v[2]=(__bf16)a.z;  v[3]=(__bf16)a.w;
  v[4]=(__bf16)b.x;  v[5]=(__bf16)b.y;  v[6]=(__bf16)b.z;  v[7]=(__bf16)b.w;
  v[8]=(__bf16)c.x;  v[9]=(__bf16)c.y;  v[10]=(__bf16)c.z; v[11]=(__bf16)c.w;
  v[12]=(__bf16)d.x; v[13]=(__bf16)d.y; v[14]=(__bf16)d.z; v[15]=(__bf16)d.w;
  return v;
}

static __device__ __forceinline__ u32 pk2(float lo, float hi) {
  unsigned short a = __builtin_bit_cast(unsigned short, (__bf16)lo);
  unsigned short b = __builtin_bit_cast(unsigned short, (__bf16)hi);
  return (u32)a | ((u32)b << 16);
}

// 16-byte global -> LDS copy: async path (ASYNCcnt-tracked, no VGPR staging)
// when the toolchain exposes it, otherwise plain load + ds_store.
static __device__ __forceinline__ void cp16_g2l(__bf16* ldst, const __bf16* gsrc) {
#if HAS_ASYNC_LDS
  __builtin_amdgcn_global_load_async_to_lds_b128(
      (__attribute__((address_space(1))) v4i*)gsrc,
      (__attribute__((address_space(3))) v4i*)ldst, 0, 0);
#else
  *(uint4*)ldst = *(const uint4*)gsrc;
#endif
}

static __device__ __forceinline__ void wait_async_lds() {
#if HAS_ASYNC_LDS
#  if HAS_WAIT_ASYNC
  __builtin_amdgcn_s_wait_asynccnt(0);
#  else
  asm volatile("s_wait_asynccnt 0x0" ::: "memory");
#  endif
#endif
}

// One wave computes a 16x64 fp32 tile: C = A(16xD) * W(D x 64 cols at c0).
// The 32xk-slice x 64-col W chunk is staged into LDS transposed as bf16 by the
// whole block, double-buffered: one barrier per k-step, staging of chunk k+1
// overlaps the 4 WMMAs of chunk k. B-fragments are batched (8x ds_load_b128,
// one wait) so the WMMAs issue back-to-back.
template<bool A_BF16>
static __device__ __forceinline__ void gemm_mainloop(const void* Abase_,
                                                     const float* W, int ldw, int c0,
                                                     __bf16* Wl, v8f acc[4]) {
  const int tid = threadIdx.x;
  const int lane = tid & 31, ln = lane & 15, laneHi = lane >> 4;
  const int aoff = laneHi ? 8 : 0;
  const int nn = tid & 63;          // output column this thread stages
  const int kkb = tid >> 6;         // 0..3

  #pragma unroll
  for (int i = 0; i < 8; ++i) {     // stage chunk 0
    int kk = kkb + i * 4;
    Wl[nn * WPAD + kk] = (__bf16)W[(size_t)kk * ldw + c0 + nn];
  }
  __syncthreads();

  for (int k0 = 0; k0 < DMODEL; k0 += 32) {
    const int cur = (k0 >> 5) & 1;
    __bf16* wcur = Wl + cur * (64 * WPAD);
    if (k0 + 32 < DMODEL) {                       // stage next chunk (overlaps)
      __bf16* wnxt = Wl + (cur ^ 1) * (64 * WPAD);
      #pragma unroll
      for (int i = 0; i < 8; ++i) {
        int kk = kkb + i * 4;
        wnxt[nn * WPAD + kk] = (__bf16)W[(size_t)(k0 + 32 + kk) * ldw + c0 + nn];
      }
    }
    v16bf af;
    if constexpr (A_BF16) af = loadA_bf16((const __bf16*)Abase_ + k0, aoff);
    else                  af = loadA_f32 ((const float*)Abase_ + k0, aoff);
    v16bf bf[4];
    #pragma unroll
    for (int nt = 0; nt < 4; ++nt)
      bf[nt] = loadA_bf16(wcur + (nt * 16 + ln) * WPAD, aoff);
    #pragma unroll
    for (int nt = 0; nt < 4; ++nt)
      acc[nt] = __builtin_amdgcn_wmma_f32_16x16x32_bf16(
          false, af, false, bf[nt], (short)0, acc[nt], false, false);
    __syncthreads();
  }
}

// Fused per-head RMSNorm (d=64) + RoPE on a wave's 16x64 C tile.
// C layout: VGPR r holds row r (lanes 0-15) / row r+8 (lanes 16-31), col = nt*16+ln.
static __device__ __forceinline__ void norm_rope(v8f acc[4], const float* g,
                                                 const int* pos, int rowBase,
                                                 int ln, int laneHi) {
  float gn[4];
  #pragma unroll
  for (int nt = 0; nt < 4; ++nt) gn[nt] = g[nt * 16 + ln];
  #pragma unroll
  for (int r = 0; r < 8; ++r) {
    float s = 0.f;
    #pragma unroll
    for (int nt = 0; nt < 4; ++nt) { float t = acc[nt][r]; s += t * t; }
    #pragma unroll
    for (int off = 1; off < 16; off <<= 1) s += __shfl_xor(s, off, 16);
    float rs = rsqrtf(s * (1.0f / 64.0f) + 1.1920929e-7f);
    #pragma unroll
    for (int nt = 0; nt < 4; ++nt) acc[nt][r] *= rs * gn[nt];
  }
  const float lb = 0.2878231366f;                 // ln(10000) * (2/64)
  float inv0 = __expf(-(float)ln * lb);           // pair index ln
  float inv1 = __expf(-(float)(ln + 16) * lb);    // pair index ln+16
  #pragma unroll
  for (int r = 0; r < 8; ++r) {
    int row = rowBase + r + (laneHi ? 8 : 0);
    float p = (float)pos[row & (SEQL - 1)];
    float a0 = p * inv0, a1 = p * inv1;
    float c0 = __cosf(a0), s0 = __sinf(a0);
    float c1 = __cosf(a1), s1 = __sinf(a1);
    float x0 = acc[0][r], x2 = acc[2][r];
    acc[0][r] = x0 * c0 - x2 * s0;
    acc[2][r] = x0 * s0 + x2 * c0;
    float x1 = acc[1][r], x3 = acc[3][r];
    acc[1][r] = x1 * c1 - x3 * s1;
    acc[3][r] = x1 * s1 + x3 * c1;
  }
}

// ---------------- Q projection: q = rope(rmsnorm(X @ Wq)) -> bf16 [B,H,L,64]
__global__ __launch_bounds__(256) void k_proj_q(const float* __restrict__ X,
                                                const float* __restrict__ Wq,
                                                const float* __restrict__ g,
                                                const int* __restrict__ pos,
                                                __bf16* __restrict__ Qout) {
  __shared__ alignas(16) __bf16 Wl[2 * 64 * WPAD];
  const int h = blockIdx.x % NHEADS, rb = blockIdx.x / NHEADS;
  const int w = threadIdx.x >> 5, lane = threadIdx.x & 31;
  const int ln = lane & 15, laneHi = lane >> 4;
  const int rowBase = rb * 128 + w * 16;
  v8f z = {}; v8f acc[4] = {z, z, z, z};
  gemm_mainloop<false>(X + (size_t)(rowBase + ln) * DMODEL, Wq, DMODEL,
                       h * DHEAD, Wl, acc);
  norm_rope(acc, g, pos, rowBase, ln, laneHi);
  #pragma unroll
  for (int nt = 0; nt < 4; ++nt)
    #pragma unroll
    for (int r = 0; r < 8; ++r) {
      int row = rowBase + r + (laneHi ? 8 : 0);
      int l = row & (SEQL - 1), b = row >> 11;
      Qout[((size_t)(b * NHEADS + h) * SEQL + l) * DHEAD + nt * 16 + ln] =
          (__bf16)acc[nt][r];
    }
}

// ---------------- KV projection: k (norm+rope) and v (stored transposed)
__global__ __launch_bounds__(256) void k_proj_kv(const float* __restrict__ X,
                                                 const float* __restrict__ Wkv,
                                                 const float* __restrict__ g,
                                                 const int* __restrict__ pos,
                                                 __bf16* __restrict__ Kout,
                                                 __bf16* __restrict__ Vt) {
  __shared__ alignas(16) __bf16 Wl[2 * 64 * WPAD];
  const int part = blockIdx.x & 1;          // 0 = K, 1 = V
  const int t = blockIdx.x >> 1;
  const int h = t % NHEADS, rb = t / NHEADS;
  const int w = threadIdx.x >> 5, lane = threadIdx.x & 31;
  const int ln = lane & 15, laneHi = lane >> 4;
  const int rowBase = rb * 128 + w * 16;
  v8f z = {}; v8f acc[4] = {z, z, z, z};
  gemm_mainloop<false>(X + (size_t)(rowBase + ln) * DMODEL, Wkv, 2 * DMODEL,
                       h * 2 * DHEAD + part * DHEAD, Wl, acc);
  if (part == 0) {
    norm_rope(acc, g, pos, rowBase, ln, laneHi);
    #pragma unroll
    for (int nt = 0; nt < 4; ++nt)
      #pragma unroll
      for (int r = 0; r < 8; ++r) {
        int row = rowBase + r + (laneHi ? 8 : 0);
        int l = row & (SEQL - 1), b = row >> 11;
        Kout[((size_t)(b * NHEADS + h) * SEQL + l) * DHEAD + nt * 16 + ln] =
            (__bf16)acc[nt][r];
      }
  } else {
    #pragma unroll
    for (int nt = 0; nt < 4; ++nt)
      #pragma unroll
      for (int r = 0; r < 8; ++r) {
        int row = rowBase + r + (laneHi ? 8 : 0);
        int l = row & (SEQL - 1), b = row >> 11;
        Vt[((size_t)(b * NHEADS + h) * DHEAD + nt * 16 + ln) * SEQL + l] =
            (__bf16)acc[nt][r];
      }
  }
}

// Stage one 32-key K tile (32x64) and V tile (64x32) into LDS; bf16 is already
// in the exact fragment layout, so this is a byte-exact 16B-chunk copy:
// K = 256 chunks, V = 256 chunks, one of each per thread.
static __device__ __forceinline__ void stage_kv(__bf16* Kl, __bf16* Vl,
                                                const __bf16* Kg, const __bf16* Vg,
                                                int kb, int tid) {
  {
    int row = tid >> 3, off = (tid & 7) * 8;                    // 32 rows x 128B
    cp16_g2l(Kl + row * KLP + off, Kg + (size_t)(kb + row) * DHEAD + off);
  }
  {
    int row = tid >> 2, off = (tid & 3) * 8;                    // 64 rows x 64B
    cp16_g2l(Vl + row * VLP + off, Vg + (size_t)row * SEQL + kb + off);
  }
}

// ---------------- Flash attention: one wave owns 16 queries, streams 2048 keys.
// K/V tiles are staged block-wide into double-buffered LDS (async if available),
// S^T = K*Q^T so scores sit in-lane per query, online softmax, O^T = V^T*P^T.
__global__ __launch_bounds__(256) void k_fa(const __bf16* __restrict__ Q,
                                            const __bf16* __restrict__ K,
                                            const __bf16* __restrict__ V,
                                            __bf16* __restrict__ Y) {
  __shared__ alignas(16) __bf16 Kl[2][32 * KLP];
  __shared__ alignas(16) __bf16 Vl[2][64 * VLP];
  const int qb = blockIdx.x & 15;
  const int bh = blockIdx.x >> 4;
  const int tid = threadIdx.x;
  const int w = tid >> 5, lane = tid & 31;
  const int ln = lane & 15, laneHi = lane >> 4;
  const int qbase = qb * 128 + w * 16;
  const int boff = laneHi ? 16 : 0;   // B-operand K split across lane halves
  const int aoff = laneHi ? 8 : 0;    // A-operand run base per lane half
  const __bf16* Qrow = Q + ((size_t)bh * SEQL + qbase + ln) * DHEAD;
  v16bf qf0 = loadA_bf16(Qrow, aoff);        // dims 0..31 slice
  v16bf qf1 = loadA_bf16(Qrow + 32, aoff);   // dims 32..63 slice
  const __bf16* Kb = K + (size_t)bh * SEQL * DHEAD;
  const __bf16* Vb = V + (size_t)bh * DHEAD * SEQL;
  v8f z = {}; v8f acc[4] = {z, z, z, z};
  float m = -1e30f, lsum = 0.f;

  stage_kv(Kl[0], Vl[0], Kb, Vb, 0, tid);
  wait_async_lds();
  __syncthreads();

  for (int kb = 0; kb < SEQL; kb += 32) {
    const int cur = (kb >> 5) & 1;
    if (kb + 32 < SEQL)                       // stage next tile (overlaps compute)
      stage_kv(Kl[cur ^ 1], Vl[cur ^ 1], Kb, Vb, kb + 32, tid);

    const __bf16* kr0 = Kl[cur] + ln * KLP;
    const __bf16* kr1 = Kl[cur] + (16 + ln) * KLP;
    v16bf a00 = loadA_bf16(kr0, aoff);
    v16bf a01 = loadA_bf16(kr0 + 32, aoff);
    v16bf a10 = loadA_bf16(kr1, aoff);
    v16bf a11 = loadA_bf16(kr1 + 32, aoff);
    v16bf va[4];                              // V fragments early: latency hides
    #pragma unroll                            // under the softmax VALU work
    for (int mt = 0; mt < 4; ++mt)
      va[mt] = load16bf(Vl[cur] + (mt * 16 + ln) * VLP + boff);

    v8f st0 = z, st1 = z;
    st0 = __builtin_amdgcn_wmma_f32_16x16x32_bf16(false, a00, false, qf0, (short)0, st0, false, false);
    st0 = __builtin_amdgcn_wmma_f32_16x16x32_bf16(false, a01, false, qf1, (short)0, st0, false, false);
    st1 = __builtin_amdgcn_wmma_f32_16x16x32_bf16(false, a10, false, qf0, (short)0, st1, false, false);
    st1 = __builtin_amdgcn_wmma_f32_16x16x32_bf16(false, a11, false, qf1, (short)0, st1, false, false);

    // online softmax per query (in-lane + partner at lane^16)
    float tmax = -1e30f;
    #pragma unroll
    for (int r = 0; r < 8; ++r) {
      float x0 = st0[r] * 0.125f, x1 = st1[r] * 0.125f;
      st0[r] = x0; st1[r] = x1;
      tmax = fmaxf(tmax, fmaxf(x0, x1));
    }
    tmax = fmaxf(tmax, __shfl_xor(tmax, 16, 32));
    float mn = fmaxf(m, tmax);
    float alpha = __expf(m - mn);
    float psum = 0.f;
    #pragma unroll
    for (int r = 0; r < 8; ++r) {
      float p0 = __expf(st0[r] - mn);
      float p1 = __expf(st1[r] - mn);
      st0[r] = p0; st1[r] = p1;
      psum += p0 + p1;
    }
    psum += __shfl_xor(psum, 16, 32);
    lsum = lsum * alpha + psum;
    m = mn;
    #pragma unroll
    for (int mt = 0; mt < 4; ++mt)
      #pragma unroll
      for (int r = 0; r < 8; ++r) acc[mt][r] *= alpha;

    // Build P^T B-operand: pack to bf16 dwords, swap tile halves with lane^16.
    u32 t0p[4], t1p[4];
    #pragma unroll
    for (int j = 0; j < 4; ++j) {
      t0p[j] = pk2(st0[2 * j], st0[2 * j + 1]);
      t1p[j] = pk2(st1[2 * j], st1[2 * j + 1]);
    }
    u32 xs[4], ys[4];
    #pragma unroll
    for (int j = 0; j < 4; ++j) {
      xs[j] = laneHi ? (u32)t0p[j] : (u32)t1p[j];
      ys[j] = __shfl_xor(xs[j], 16, 32);
    }
    VB16 pb;
    if (laneHi) {
      pb.u[0] = make_uint4(ys[0], ys[1], ys[2], ys[3]);     // keys 16..23
      pb.u[1] = make_uint4(t1p[0], t1p[1], t1p[2], t1p[3]); // keys 24..31
    } else {
      pb.u[0] = make_uint4(t0p[0], t0p[1], t0p[2], t0p[3]); // keys 0..7
      pb.u[1] = make_uint4(ys[0], ys[1], ys[2], ys[3]);     // keys 8..15
    }

    // O^T += V^T(16 dims x 32 keys) * P^T(32 keys x 16 queries)
    #pragma unroll
    for (int mt = 0; mt < 4; ++mt)
      acc[mt] = __builtin_amdgcn_wmma_f32_16x16x32_bf16(
          false, va[mt], false, pb.v, (short)0, acc[mt], false, false);

    wait_async_lds();
    __syncthreads();
  }

  float invl = 1.0f / lsum;
  const int b = bh / NHEADS, h = bh % NHEADS;
  #pragma unroll
  for (int mt = 0; mt < 4; ++mt)
    #pragma unroll
    for (int r = 0; r < 8; ++r) {
      int dim = mt * 16 + r + (laneHi ? 8 : 0);
      int q = qbase + ln;
      Y[((size_t)b * SEQL + q) * DMODEL + h * DHEAD + dim] =
          (__bf16)(acc[mt][r] * invl);
    }
}

// ---------------- Output projection: out = Y @ Wout (fp32 result)
__global__ __launch_bounds__(256) void k_out(const __bf16* __restrict__ Yb,
                                             const float* __restrict__ Wout,
                                             float* __restrict__ Out) {
  __shared__ alignas(16) __bf16 Wl[2 * 64 * WPAD];
  const int cb = blockIdx.x & 15, rb = blockIdx.x >> 4;
  const int w = threadIdx.x >> 5, lane = threadIdx.x & 31;
  const int ln = lane & 15, laneHi = lane >> 4;
  const int rowBase = rb * 128 + w * 16;
  v8f z = {}; v8f acc[4] = {z, z, z, z};
  gemm_mainloop<true>(Yb + (size_t)(rowBase + ln) * DMODEL, Wout, DMODEL,
                      cb * 64, Wl, acc);
  #pragma unroll
  for (int nt = 0; nt < 4; ++nt)
    #pragma unroll
    for (int r = 0; r < 8; ++r) {
      int row = rowBase + r + (laneHi ? 8 : 0);
      Out[(size_t)row * DMODEL + cb * 64 + nt * 16 + ln] = acc[nt][r];
    }
}

extern "C" void kernel_launch(void* const* d_in, const int* in_sizes, int n_in,
                              void* d_out, int out_size, void* d_ws, size_t ws_size,
                              hipStream_t stream) {
  const float* queries = (const float*)d_in[0];
  const float* kv      = (const float*)d_in[1];
  const float* Wq      = (const float*)d_in[2];
  const float* Wkv     = (const float*)d_in[3];
  const float* Wout    = (const float*)d_in[4];
  const float* g_q     = (const float*)d_in[5];
  const float* g_k     = (const float*)d_in[6];
  const int*   pos_q   = (const int*)d_in[7];
  const int*   pos_k   = (const int*)d_in[8];
  (void)in_sizes; (void)n_in; (void)out_size; (void)ws_size;

  __bf16* ws = (__bf16*)d_ws;
  const size_t SZ = (size_t)BATCH * NHEADS * SEQL * DHEAD;  // 8.39M elems
  __bf16* Qb  = ws;
  __bf16* Kb  = ws + SZ;
  __bf16* Vtb = ws + 2 * SZ;
  __bf16* Yb  = ws + 3 * SZ;

  const int rowBlocks = (BATCH * SEQL) / 128;               // 64
  k_proj_q <<<rowBlocks * NHEADS,          256, 0, stream>>>(queries, Wq,  g_q, pos_q, Qb);
  k_proj_kv<<<rowBlocks * NHEADS * 2,      256, 0, stream>>>(kv,      Wkv, g_k, pos_k, Kb, Vtb);
  k_fa     <<<BATCH * NHEADS * (SEQL/128), 256, 0, stream>>>(Qb, Kb, Vtb, Yb);
  k_out    <<<rowBlocks * (DMODEL/64),     256, 0, stream>>>(Yb, Wout, (float*)d_out);
}
